// CausalSelfAttention_84628035600666
// MI455X (gfx1250) — matrix operand order repaired and checked by reading, not compiled, not run on hardware
//
#include <hip/hip_runtime.h>
#include <hip/hip_bf16.h>

typedef __bf16 bf16_t;
typedef __attribute__((ext_vector_type(16))) __bf16 v16bf;
typedef __attribute__((ext_vector_type(8)))  __bf16 v8bf;
typedef __attribute__((ext_vector_type(8)))  float  v8f;
typedef __attribute__((ext_vector_type(4)))  unsigned int u32x4;
typedef __attribute__((ext_vector_type(8)))  int i32x8;
typedef __attribute__((ext_vector_type(4)))  int i32x4;

#define B_   2
#define T_   2048
#define C_   1024
#define H_   16
#define HD_  64
#define M_TOT (B_ * T_)   // 4096

// ---------------- f32 -> bf16 conversion ----------------
__global__ void cvt_f32_bf16(const float* __restrict__ in, bf16_t* __restrict__ out, int n) {
  int i = blockIdx.x * blockDim.x + threadIdx.x;
  if (i < n) out[i] = (bf16_t)in[i];
}

// A-fragment (16x32 bf16, contiguous K pattern): elems 0..15 = K k0..k0+15
__device__ __forceinline__ v16bf ld_a_frag(const bf16_t* p, int k0) {
  union { v16bf v; v8bf h[2]; } u;
  u.h[0] = *(const v8bf*)(p + k0);
  u.h[1] = *(const v8bf*)(p + k0 + 8);
  return u.v;
}
// B-fragment (32x16 bf16): lane half holds 16 contiguous K values
__device__ __forceinline__ v16bf ld_b_frag(const bf16_t* p) {
  union { v16bf v; v8bf h[2]; } u;
  u.h[0] = *(const v8bf*)(p);
  u.h[1] = *(const v8bf*)(p + 8);
  return u.v;
}
__device__ __forceinline__ v8f wmma_bf16(v16bf a, v16bf b, v8f c) {
  return __builtin_amdgcn_wmma_f32_16x16x32_bf16(false, a, false, b, (short)0, c, false, false);
}

// ---------------- TDM: 2D tile load Global -> LDS (ISA 08_async_tensor §8) ----------------
// Loads tile_d1 rows of tile_d0 elements (bf16, data_size=2B) starting at gaddr, row stride
// stride0 elements; LDS destination gets pad_amount DWORD padding after every pad_interval
// DWORDs (codes per D# group1: interval 2^(c+1) DW, amount c+1 DW).
__device__ __forceinline__ void tdm_load_2d(const void* lds_dst, const void* gaddr,
                                            unsigned td0, unsigned td1,
                                            unsigned tile_d0, unsigned tile_d1,
                                            unsigned long long stride0,
                                            unsigned padi, unsigned pada) {
  unsigned long long ga = (unsigned long long)(uintptr_t)gaddr;
  unsigned lds_off = (unsigned)(uintptr_t)lds_dst;   // low 32 bits of flat addr = LDS offset
  u32x4 g0;
  g0[0] = 1u;                                        // count=1, user mode, no gather
  g0[1] = lds_off;                                   // lds_addr [63:32]
  g0[2] = (unsigned)ga;                              // global_addr[31:0]
  g0[3] = (unsigned)((ga >> 32) & 0x1FFFFFFu) | (2u << 30);  // addr[56:32] | type=2
  i32x8 g1;
  g1[0] = (int)((1u << 16) | (1u << 20) | (padi << 22) | (pada << 25)); // data_size=2B, pad_en
  g1[1] = (int)((td0 & 0xFFFFu) << 16);              // tensor_dim0[15:0] @ bits 63:48
  g1[2] = (int)((td0 >> 16) | ((td1 & 0xFFFFu) << 16));
  g1[3] = (int)((td1 >> 16) | (tile_d0 << 16));      // tile_dim0 @ bits 127:112
  g1[4] = (int)tile_d1;                              // tile_dim1; tile_dim2=0
  g1[5] = (int)(unsigned)stride0;                    // tensor_dim0_stride[31:0]
  g1[6] = (int)(unsigned)(stride0 >> 32);            // stride[47:32]; dim1_stride=0
  g1[7] = 0;
  i32x4 z4 = {0, 0, 0, 0};
  i32x8 z8 = {0, 0, 0, 0, 0, 0, 0, 0};
  __builtin_amdgcn_tensor_load_to_lds(g0, g1, z4, z4, z8, 0);
}

// ---------------- tiled GEMM: D = A @ W^T + bias ----------------
// MODE 0: QKV -> write q (scaled 1/8, [BH][T][HD]), k ([BH][T][HD]), vt ([BH][HD][T]) as bf16
// MODE 1: proj -> write f32 out [M][C]
template <int MODE>
__global__ __launch_bounds__(256) void gemm_bf16_kernel(
    const bf16_t* __restrict__ A, const bf16_t* __restrict__ W,
    const float* __restrict__ bias,
    bf16_t* __restrict__ q, bf16_t* __restrict__ kbuf, bf16_t* __restrict__ vt,
    float* __restrict__ outf, int K) {
  constexpr int BM = 128, BN = 64, BK = 32, LDT = 40;  // rows: 16 DW data + 4 DW pad
  __shared__ bf16_t sA[2][BM * LDT];
  __shared__ bf16_t sB[2][BN * LDT];

  const int tid  = threadIdx.x;
  const int lane = tid & 31;
  const int wave = tid >> 5;
  const int wm   = (wave >> 1) * 32;
  const int wn   = (wave & 1) * 32;
  const int l16  = lane & 15;
  const int k0   = (lane >> 4) * 16;
  const int kc   = (lane >> 4) * 16;
  const int hi   = (lane >> 4) * 8;

  const int m0 = blockIdx.y * BM;
  const int n0 = blockIdx.x * BN;

  v8f acc[2][2] = {};

  // prologue: TDM-stage first K tile into buffer 0 (wave0: A, wave1: B)
  if (wave == 0)
    tdm_load_2d(&sA[0][0], A + (size_t)m0 * K, K, M_TOT, BK, BM, K, 3, 3);
  else if (wave == 1)
    tdm_load_2d(&sB[0][0], W + (size_t)n0 * K, K, 4096, BK, BN, K, 3, 3);

  for (int kb = 0; kb < K; kb += BK) {
    const int p = (kb >> 5) & 1;
    const bool more = (kb + BK) < K;
    __syncthreads();   // all readers of buffer 1-p (iter-1 compute) are done
    if (wave == 0) {
      if (more) {
        tdm_load_2d(&sA[1 - p][0], A + (size_t)m0 * K + kb + BK, K, M_TOT, BK, BM, K, 3, 3);
        __builtin_amdgcn_s_wait_tensorcnt((short)1);   // buffer p complete
      } else {
        __builtin_amdgcn_s_wait_tensorcnt((short)0);
      }
    } else if (wave == 1) {
      if (more) {
        tdm_load_2d(&sB[1 - p][0], W + (size_t)n0 * K + kb + BK, K, 4096, BK, BN, K, 3, 3);
        __builtin_amdgcn_s_wait_tensorcnt((short)1);
      } else {
        __builtin_amdgcn_s_wait_tensorcnt((short)0);
      }
    }
    __syncthreads();   // tile p visible to all waves

    v16bf af[2], bf[2];
    #pragma unroll
    for (int mi = 0; mi < 2; ++mi)
      af[mi] = ld_a_frag(&sA[p][(wm + mi * 16 + l16) * LDT], k0);
    #pragma unroll
    for (int ni = 0; ni < 2; ++ni)
      bf[ni] = ld_b_frag(&sB[p][(wn + ni * 16 + l16) * LDT + kc]);
    #pragma unroll
    for (int mi = 0; mi < 2; ++mi)
      #pragma unroll
      for (int ni = 0; ni < 2; ++ni)
        acc[mi][ni] = wmma_bf16(af[mi], bf[ni], acc[mi][ni]);
  }

  // C/D layout: lane holds col N=l16; VGPR r holds row M = hi + r
  #pragma unroll
  for (int mi = 0; mi < 2; ++mi) {
    #pragma unroll
    for (int ni = 0; ni < 2; ++ni) {
      const int gn = n0 + wn + ni * 16 + l16;
      const float bv = bias[gn];
      #pragma unroll
      for (int r = 0; r < 8; ++r) {
        const int gm = m0 + wm + mi * 16 + hi + r;
        const float val = acc[mi][ni][r] + bv;
        if (MODE == 0) {
          const int which = gn >> 10;             // 0=q,1=k,2=v
          const int cc = gn & (C_ - 1);
          const int h = cc >> 6, d = cc & (HD_ - 1);
          const int b = gm >> 11, t = gm & (T_ - 1);
          const size_t bh = (size_t)b * H_ + h;
          if (which == 0)      q[(bh * T_ + t) * HD_ + d]    = (bf16_t)(val * 0.125f); // 1/sqrt(64)
          else if (which == 1) kbuf[(bh * T_ + t) * HD_ + d] = (bf16_t)val;
          else                 vt[(bh * HD_ + d) * T_ + t]   = (bf16_t)val;            // transposed V
        } else {
          outf[(size_t)gm * C_ + gn] = val;
        }
      }
    }
  }
}

// ---------------- flash attention ----------------
// grid: (T/64, B*H), block 128 (4 waves x 16 queries). KV tiles of 32 keys.
__global__ __launch_bounds__(128) void attn_kernel(
    const bf16_t* __restrict__ q, const bf16_t* __restrict__ kk,
    const bf16_t* __restrict__ vt, bf16_t* __restrict__ y) {
  constexpr int LDK = 72, LDV = 40, LDP = 40;
  __shared__ bf16_t sK[32 * LDK];       // 32 keys x 64 d  (rows: 32 DW + 4 DW pad)
  __shared__ bf16_t sV[HD_ * LDV];      // 64 d x 32 keys  (rows: 16 DW + 4 DW pad)
  __shared__ bf16_t sP[4 * 16 * LDP];   // per-wave P (16 q x 32 keys)

  const int tid  = threadIdx.x;
  const int lane = tid & 31;
  const int wave = tid >> 5;
  const int l16  = lane & 15;
  const int k0   = (lane >> 4) * 16;
  const int kc   = (lane >> 4) * 16;
  const int hi   = (lane >> 4) * 8;

  const int bh    = blockIdx.y;
  const int b     = bh >> 4, h = bh & (H_ - 1);
  const int qBase = blockIdx.x * 64;
  const int qRow  = qBase + wave * 16 + l16;

  // Q A-fragments (pre-scaled by 1/sqrt(HD) at QKV time)
  const bf16_t* qp = q + ((size_t)bh * T_ + qRow) * HD_;
  const v16bf qf0 = ld_a_frag(qp, k0);        // d 0..31
  const v16bf qf1 = ld_a_frag(qp + 32, k0);   // d 32..63

  float mrun[8], lrun[8];
  v8f o[4] = {};
  #pragma unroll
  for (int r = 0; r < 8; ++r) { mrun[r] = -1e30f; lrun[r] = 0.f; }

  const int kvEnd = qBase + 64;                // causal bound (block-uniform)
  for (int kv = 0; kv < kvEnd; kv += 32) {
    __syncthreads();   // previous iteration's sK/sV readers done
    if (wave == 0) {   // K tile: 32 rows x 64 d, row stride HD
      tdm_load_2d(&sK[0], kk + ((size_t)bh * T_ + kv) * HD_, HD_, T_, HD_, 32, HD_, 4, 3);
      __builtin_amdgcn_s_wait_tensorcnt((short)0);
    } else if (wave == 1) {  // V^T tile: 64 rows x 32 keys, row stride T
      tdm_load_2d(&sV[0], vt + (size_t)bh * HD_ * T_ + kv, T_, HD_, 32, HD_, T_, 3, 3);
      __builtin_amdgcn_s_wait_tensorcnt((short)0);
    }
    __syncthreads();

    // S = Q K^T  (two 16-key N-tiles, two K-steps over d)
    v8f s[2] = {};
    #pragma unroll
    for (int ni = 0; ni < 2; ++ni) {
      const bf16_t* krow = &sK[(ni * 16 + l16) * LDK];
      s[ni] = wmma_bf16(qf0, ld_b_frag(krow + kc), s[ni]);
      s[ni] = wmma_bf16(qf1, ld_b_frag(krow + 32 + kc), s[ni]);
    }

    // causal mask (only on diagonal-touching tiles; wave-uniform branch)
    if (kv + 31 > qBase + wave * 16) {
      #pragma unroll
      for (int ni = 0; ni < 2; ++ni) {
        const int kj = kv + ni * 16 + l16;
        #pragma unroll
        for (int r = 0; r < 8; ++r)
          if (kj > qBase + wave * 16 + hi + r) s[ni][r] = -1e30f;
      }
    }

    // online softmax; row stats shared across 16-lane half-groups via shfl_xor
    float p0[8], p1[8];
    #pragma unroll
    for (int r = 0; r < 8; ++r) {
      float v = fmaxf(s[0][r], s[1][r]);
      #pragma unroll
      for (int off = 1; off < 16; off <<= 1) v = fmaxf(v, __shfl_xor(v, off, 32));
      const float mnew = fmaxf(mrun[r], v);
      const float sc = __expf(mrun[r] - mnew);
      mrun[r] = mnew;
      lrun[r] *= sc;
      #pragma unroll
      for (int di = 0; di < 4; ++di) o[di][r] *= sc;
      p0[r] = __expf(s[0][r] - mnew);
      p1[r] = __expf(s[1][r] - mnew);
      float ps = p0[r] + p1[r];
      #pragma unroll
      for (int off = 1; off < 16; off <<= 1) ps += __shfl_xor(ps, off, 32);
      lrun[r] += ps;
    }

    // C-layout -> A-layout via per-wave LDS round trip
    bf16_t* sp = &sP[wave * 16 * LDP];
    #pragma unroll
    for (int r = 0; r < 8; ++r) {
      sp[(hi + r) * LDP + l16]      = (bf16_t)p0[r];
      sp[(hi + r) * LDP + 16 + l16] = (bf16_t)p1[r];
    }
    __syncthreads();

    // O += P @ V  (K = 32 keys, four 16-wide d tiles)
    const v16bf pf = ld_a_frag(&sp[l16 * LDP], k0);
    #pragma unroll
    for (int di = 0; di < 4; ++di)
      o[di] = wmma_bf16(pf, ld_b_frag(&sV[(di * 16 + l16) * LDV + kc]), o[di]);
  }

  // normalize, store y as [B,T,C] bf16 for the projection GEMM
  #pragma unroll
  for (int r = 0; r < 8; ++r) {
    const float inv = 1.0f / lrun[r];
    const int t = qBase + wave * 16 + hi + r;
    const size_t base = ((size_t)b * T_ + t) * C_ + h * HD_;
    #pragma unroll
    for (int di = 0; di < 4; ++di)
      y[base + di * 16 + l16] = (bf16_t)(o[di][r] * inv);
  }
}

extern "C" void kernel_launch(void* const* d_in, const int* in_sizes, int n_in,
                              void* d_out, int out_size, void* d_ws, size_t ws_size,
                              hipStream_t stream) {
  (void)in_sizes; (void)n_in; (void)out_size; (void)ws_size;
  const float* x      = (const float*)d_in[0];
  const float* w_attn = (const float*)d_in[1];
  const float* b_attn = (const float*)d_in[2];
  const float* w_proj = (const float*)d_in[3];
  const float* b_proj = (const float*)d_in[4];
  float* out = (float*)d_out;

  bf16_t* ws = (bf16_t*)d_ws;
  size_t off = 0;
  bf16_t* xb  = ws + off; off += (size_t)M_TOT * C_;
  bf16_t* wab = ws + off; off += (size_t)3 * C_ * C_;
  bf16_t* wpb = ws + off; off += (size_t)C_ * C_;
  bf16_t* qb  = ws + off; off += (size_t)B_ * H_ * T_ * HD_;
  bf16_t* kb  = ws + off; off += (size_t)B_ * H_ * T_ * HD_;
  bf16_t* vtb = ws + off; off += (size_t)B_ * H_ * T_ * HD_;
  bf16_t* yb  = ws + off;

  const int nx = M_TOT * C_;
  cvt_f32_bf16<<<(nx + 255) / 256, 256, 0, stream>>>(x, xb, nx);
  const int nwa = 3 * C_ * C_;
  cvt_f32_bf16<<<(nwa + 255) / 256, 256, 0, stream>>>(w_attn, wab, nwa);
  const int nwp = C_ * C_;
  cvt_f32_bf16<<<(nwp + 255) / 256, 256, 0, stream>>>(w_proj, wpb, nwp);

  dim3 g1(3 * C_ / 64, M_TOT / 128);   // (48, 32)
  gemm_bf16_kernel<0><<<g1, 256, 0, stream>>>(xb, wab, b_attn, qb, kb, vtb, nullptr, C_);

  dim3 g2(T_ / 64, B_ * H_);           // (32, 32)
  attn_kernel<<<g2, 128, 0, stream>>>(qb, kb, vtb, yb);

  dim3 g3(C_ / 64, M_TOT / 128);       // (16, 32)
  gemm_bf16_kernel<1><<<g3, 256, 0, stream>>>(yb, wpb, b_proj, nullptr, nullptr, nullptr, out, C_);
}
